// RipsNet_25297357373836
// MI455X (gfx1250) — matrix operand order repaired and checked by reading, not compiled, not run on hardware
//
#include <hip/hip_runtime.h>

// ---------------------------------------------------------------------------
// Types for CDNA5 WMMA (wave32): D[16x16 f32] = A[16x32 bf16] * B[32x16 bf16] + C
// ---------------------------------------------------------------------------
typedef __attribute__((ext_vector_type(16))) __bf16 v16bf;
typedef __attribute__((ext_vector_type(8)))  float  v8f;

union Vu {
  v16bf v;
  uint4 u[2];
  __bf16 s[16];
};

__device__ inline v8f vzero() {
  v8f c;
#pragma unroll
  for (int i = 0; i < 8; ++i) c[i] = 0.f;
  return c;
}

__device__ inline v8f wmma_bf16(v16bf a, v16bf b, v8f c) {
  // (neg_a, A, neg_b, B, c_mod, C, reuse_a, reuse_b)
  return __builtin_amdgcn_wmma_f32_16x16x32_bf16(false, a, false, b, (short)0, c,
                                                 false, false);
}

// A operand (16x32 bf16) from row-major [rows][rowlen] bf16.
// Lanes 0-15: V0-3 = K[k..k+7], V4-7 = K[k+16..k+23]; lanes 16-31: +8 elems.
__device__ inline v16bf load_a(const __bf16* base, int rowlen, int tile,
                               int kstart, int lane) {
  int row = tile * 16 + (lane & 15);
  const __bf16* p = base + (size_t)row * rowlen + kstart + ((lane >> 4) << 3);
  Vu r;
  r.u[0] = *(const uint4*)p;
  r.u[1] = *(const uint4*)(p + 16);
  return r.v;
}

// B operand (32x16 bf16); column n stored as row n of transposed weights
// Wt[npad][rowlen]. Lanes 0-15: V0-3 = K[k..k+7], V4-7 = K[k+16..k+23]; lanes 16-31: +8 elems.
__device__ inline v16bf load_b(const __bf16* base, int rowlen, int tile,
                               int kstart, int lane) {
  int row = tile * 16 + (lane & 15);
  const __bf16* p = base + (size_t)row * rowlen + kstart + ((lane >> 4) << 3);
  Vu r;
  r.u[0] = *(const uint4*)p;
  r.u[1] = *(const uint4*)(p + 16);
  return r.v;
}

// ---------------------------------------------------------------------------
// Prep kernels
// ---------------------------------------------------------------------------
__global__ void zero_f32(float* p, int n) {
  int i = blockIdx.x * 256 + threadIdx.x;
  if (i < n) p[i] = 0.f;
}

// dst[n][k] (bf16, [npad][kpad]) = src[k][n] (f32, [din][dout]), zero padded
__global__ void transpose_w(const float* __restrict__ src,
                            __bf16* __restrict__ dst, int din, int dout,
                            int kpad, int npad) {
  int idx = blockIdx.x * 256 + threadIdx.x;
  if (idx >= npad * kpad) return;
  int n = idx / kpad, k = idx % kpad;
  float v = (k < din && n < dout) ? src[(size_t)k * dout + n] : 0.f;
  dst[idx] = (__bf16)v;
}

// Per-segment point counts; segment_ids are sorted -> run-length + atomics.
__global__ void counts_kernel(const long long* __restrict__ seg,
                              float* __restrict__ counts) {
  int t = blockIdx.x * 256 + threadIdx.x;  // 16384 threads x 64 points
  long long base = (long long)t * 64;
  int cs = (int)seg[base];
  float acc = 1.f;
  for (int i = 1; i < 64; ++i) {
    int s = (int)seg[base + i];
    if (s == cs) acc += 1.f;
    else { atomicAdd(&counts[cs], acc); cs = s; acc = 1.f; }
  }
  atomicAdd(&counts[cs], acc);
}

// ---------------------------------------------------------------------------
// Fused phi_1 (3->32->64->128, ReLU) + segment-sum.  512 blocks x 8 waves.
// Each wave: 256 contiguous points = 16 WMMA groups of 16 points.
// ---------------------------------------------------------------------------
__global__ __launch_bounds__(256) void phi1_kernel(
    const float* __restrict__ points, const long long* __restrict__ seg,
    const __bf16* __restrict__ gW0, const __bf16* __restrict__ gW1,
    const __bf16* __restrict__ gW2, const float* __restrict__ gB0,
    const float* __restrict__ gB1, const float* __restrict__ gB2,
    float* __restrict__ sums) {
  __shared__ __align__(16) __bf16 sW0[32 * 32];
  __shared__ __align__(16) __bf16 sW1[64 * 32];
  __shared__ __align__(16) __bf16 sW2[128 * 64];
  __shared__ float sB0[32], sB1[64], sB2[128];
  __shared__ __align__(16) __bf16 act1[8][16 * 32];
  __shared__ __align__(16) __bf16 act2[8][16 * 64];

  for (int i = threadIdx.x; i < 32 * 32; i += 256) sW0[i] = gW0[i];
  for (int i = threadIdx.x; i < 64 * 32; i += 256) sW1[i] = gW1[i];
  for (int i = threadIdx.x; i < 128 * 64; i += 256) sW2[i] = gW2[i];
  if (threadIdx.x < 32) sB0[threadIdx.x] = gB0[threadIdx.x];
  if (threadIdx.x < 64) sB1[threadIdx.x] = gB1[threadIdx.x];
  if (threadIdx.x < 128) sB2[threadIdx.x] = gB2[threadIdx.x];
  __syncthreads();

  const int wave = threadIdx.x >> 5;
  const int lane = threadIdx.x & 31;
  const int l16 = lane & 15;
  const int half = lane >> 4;
  const int mbase = half << 3;

  __bf16* a1buf = act1[wave];
  __bf16* a2buf = act2[wave];
  const int pbase0 = blockIdx.x * 2048 + wave * 256;

  for (int g = 0; g < 16; ++g) {
    const int pbase = pbase0 + g * 16;

    // Prefetch next group's stream data (speculative; OOB silently dropped).
    __builtin_prefetch(points + (size_t)(pbase + 16 + l16) * 3, 0, 0);
    __builtin_prefetch(seg + pbase + 16 + mbase, 0, 0);

    // ---- layer 1: A built from raw points, K padded 3 -> 32 ----
    Vu a1;
    a1.u[0] = make_uint4(0, 0, 0, 0);
    a1.u[1] = make_uint4(0, 0, 0, 0);
    if (half == 0) {  // lanes 0-15 hold K=0..7 in V0-3; only K=0,1,2 nonzero
      const float* pp = points + (size_t)(pbase + l16) * 3;
      a1.s[0] = (__bf16)pp[0];
      a1.s[1] = (__bf16)pp[1];
      a1.s[2] = (__bf16)pp[2];
    }
    v8f c0 = wmma_bf16(a1.v, load_b(sW0, 32, 0, 0, lane), vzero());
    v8f c1 = wmma_bf16(a1.v, load_b(sW0, 32, 1, 0, lane), vzero());
    float b0a = sB0[l16], b0b = sB0[16 + l16];
#pragma unroll
    for (int r = 0; r < 8; ++r) {
      int m = mbase + r;
      a1buf[m * 32 + l16] = (__bf16)fmaxf(c0[r] + b0a, 0.f);
      a1buf[m * 32 + 16 + l16] = (__bf16)fmaxf(c1[r] + b0b, 0.f);
    }

    // ---- layer 2: 32 -> 64 ----
    v16bf a2 = load_a(a1buf, 32, 0, 0, lane);
    v8f d[4];
#pragma unroll
    for (int n = 0; n < 4; ++n)
      d[n] = wmma_bf16(a2, load_b(sW1, 32, n, 0, lane), vzero());
#pragma unroll
    for (int n = 0; n < 4; ++n) {
      float bb = sB1[n * 16 + l16];
#pragma unroll
      for (int r = 0; r < 8; ++r) {
        int m = mbase + r;
        a2buf[m * 64 + n * 16 + l16] = (__bf16)fmaxf(d[n][r] + bb, 0.f);
      }
    }

    // ---- layer 3: 64 -> 128 (2 K-steps) ----
    v16bf a3k0 = load_a(a2buf, 64, 0, 0, lane);
    v16bf a3k1 = load_a(a2buf, 64, 0, 32, lane);
    v8f e[8];
#pragma unroll
    for (int n = 0; n < 8; ++n) {
      v8f acc = wmma_bf16(a3k0, load_b(sW2, 64, n, 0, lane), vzero());
      e[n] = wmma_bf16(a3k1, load_b(sW2, 64, n, 32, lane), acc);
    }

    // ---- segment-sum with run-length compaction (segment_ids sorted) ----
    int sg[8];
#pragma unroll
    for (int r = 0; r < 8; ++r) sg[r] = (int)seg[pbase + mbase + r];
#pragma unroll
    for (int n = 0; n < 8; ++n) {
      float bb = sB2[n * 16 + l16];
      int f = n * 16 + l16;
      float acc = fmaxf(e[n][0] + bb, 0.f);
      int cs = sg[0];
#pragma unroll
      for (int r = 1; r < 8; ++r) {
        float v = fmaxf(e[n][r] + bb, 0.f);
        if (sg[r] == cs) acc += v;
        else { atomicAdd(&sums[(size_t)cs * 128 + f], acc); cs = sg[r]; acc = v; }
      }
      atomicAdd(&sums[(size_t)cs * 128 + f], acc);
    }
  }
}

// pooled = sums / max(counts,1), emitted as bf16 row-major [1024][128]
__global__ void pool_kernel(const float* __restrict__ sums,
                            const float* __restrict__ counts,
                            __bf16* __restrict__ pooledbf) {
  int i = blockIdx.x * 256 + threadIdx.x;  // 131072
  int row = i >> 7;
  float c = counts[row];
  c = c < 1.f ? 1.f : c;
  pooledbf[i] = (__bf16)(sums[i] / c);
}

// ---------------------------------------------------------------------------
// phi_2 GEMMs. One (mtile, ntile) 16x16 output tile per wave, 8 waves/block.
// ---------------------------------------------------------------------------
__global__ __launch_bounds__(256) void gemm_bf16_relu(
    const __bf16* __restrict__ A, const __bf16* __restrict__ Bw,
    const float* __restrict__ bias, __bf16* __restrict__ out, int K, int Nt) {
  const int wave = threadIdx.x >> 5, lane = threadIdx.x & 31;
  const int id = blockIdx.x * 8 + wave;
  const int mt = id / Nt, nt = id % Nt;
  const int l16 = lane & 15, mbase = (lane >> 4) << 3;
  v8f c = vzero();
  for (int k = 0; k < K; k += 32)
    c = wmma_bf16(load_a(A, K, mt, k, lane), load_b(Bw, K, nt, k, lane), c);
  const int Nw = Nt * 16;
  const int col = nt * 16 + l16;
  const float bb = bias[col];
#pragma unroll
  for (int r = 0; r < 8; ++r)
    out[(size_t)(mt * 16 + mbase + r) * Nw + col] =
        (__bf16)fmaxf(c[r] + bb, 0.f);
}

__global__ __launch_bounds__(256) void gemm_f32_out(
    const __bf16* __restrict__ A, const __bf16* __restrict__ Bw,
    const float* __restrict__ bias, float* __restrict__ out, int K, int Nt,
    int Nreal) {
  const int wave = threadIdx.x >> 5, lane = threadIdx.x & 31;
  const int id = blockIdx.x * 8 + wave;
  const int mt = id / Nt, nt = id % Nt;
  const int l16 = lane & 15, mbase = (lane >> 4) << 3;
  v8f c = vzero();
  for (int k = 0; k < K; k += 32) {
    // Hide HBM/L2 latency of the next K-step of streamed B rows.
    __builtin_prefetch(Bw + (size_t)(nt * 16 + l16) * K + k + 32, 0, 0);
    c = wmma_bf16(load_a(A, K, mt, k, lane), load_b(Bw, K, nt, k, lane), c);
  }
  const int col = nt * 16 + l16;
  if (col < Nreal) {
    const float bb = bias[col];
#pragma unroll
    for (int r = 0; r < 8; ++r)
      out[(size_t)(mt * 16 + mbase + r) * Nreal + col] = c[r] + bb;
  }
}

// ---------------------------------------------------------------------------
// Host-side launcher
// ---------------------------------------------------------------------------
extern "C" void kernel_launch(void* const* d_in, const int* in_sizes, int n_in,
                              void* d_out, int out_size, void* d_ws,
                              size_t ws_size, hipStream_t stream) {
  (void)in_sizes; (void)n_in; (void)out_size; (void)ws_size;
  const float* points = (const float*)d_in[0];
  const long long* seg = (const long long*)d_in[1];
  const float* W1_0 = (const float*)d_in[3];
  const float* b1_0 = (const float*)d_in[4];
  const float* W1_1 = (const float*)d_in[5];
  const float* b1_1 = (const float*)d_in[6];
  const float* W1_2 = (const float*)d_in[7];
  const float* b1_2 = (const float*)d_in[8];
  const float* W2_0 = (const float*)d_in[9];
  const float* b2_0 = (const float*)d_in[10];
  const float* W2_1 = (const float*)d_in[11];
  const float* b2_1 = (const float*)d_in[12];
  float* out = (float*)d_out;

  char* ws = (char*)d_ws;
  float* sums = (float*)(ws + 0);               // 1024*128 f32
  float* counts = (float*)(ws + 524288);        // 1024 f32
  __bf16* w1t0 = (__bf16*)(ws + 528384);        // [32][32]
  __bf16* w1t1 = (__bf16*)(ws + 530432);        // [64][32]
  __bf16* w1t2 = (__bf16*)(ws + 534528);        // [128][64]
  __bf16* w2t0 = (__bf16*)(ws + 550912);        // [256][128]
  __bf16* w2t1 = (__bf16*)(ws + 616448);        // [2512][256]
  __bf16* pooledbf = (__bf16*)(ws + 1902592);   // [1024][128]
  __bf16* h2bf = (__bf16*)(ws + 2164736);       // [1024][256]

  zero_f32<<<516, 256, 0, stream>>>(sums, 132096);  // sums + counts (contiguous)
  transpose_w<<<4, 256, 0, stream>>>(W1_0, w1t0, 3, 32, 32, 32);
  transpose_w<<<8, 256, 0, stream>>>(W1_1, w1t1, 32, 64, 32, 64);
  transpose_w<<<32, 256, 0, stream>>>(W1_2, w1t2, 64, 128, 64, 128);
  transpose_w<<<128, 256, 0, stream>>>(W2_0, w2t0, 128, 256, 128, 256);
  transpose_w<<<2512, 256, 0, stream>>>(W2_1, w2t1, 256, 2500, 256, 2512);
  counts_kernel<<<64, 256, 0, stream>>>(seg, counts);
  phi1_kernel<<<512, 256, 0, stream>>>(points, seg, w1t0, w1t1, w1t2, b1_0,
                                       b1_1, b1_2, sums);
  pool_kernel<<<512, 256, 0, stream>>>(sums, counts, pooledbf);
  gemm_bf16_relu<<<128, 256, 0, stream>>>(pooledbf, w2t0, b2_0, h2bf, 128, 16);
  gemm_f32_out<<<1256, 256, 0, stream>>>(h2bf, w2t1, b2_1, out, 256, 157, 2500);
}
